// GCN_w_position_embedding_56882546868899
// MI455X (gfx1250) — compile-verified, matrix operand order checked by reading, not run on hardware
//
#include <hip/hip_runtime.h>
#include <hip/hip_bf16.h>
#include <math.h>

// ---------------------------------------------------------------------------
// GCN with position embedding, MI455X (gfx1250) version.
// All GEMMs run on v_wmma_f32_16x16x32_bf16 (wave32 WMMA, f32 accumulate),
// with double-buffered LDS tile staging via async global->LDS copies
// (ASYNCcnt).  Wave tile 32x64 (8 WMMAs per stage), block tile 128x64.
// Softmax / LayerNorm / masking stay in f32.
// ---------------------------------------------------------------------------

typedef __bf16 bf16_t;
typedef __attribute__((ext_vector_type(16))) __bf16 v16bf;
typedef __attribute__((ext_vector_type(8)))  __bf16 v8bf;
typedef __attribute__((ext_vector_type(8)))  float  v8f;
typedef __attribute__((ext_vector_type(4)))  int    v4i;

#define NBOX   4096
#define NFG    1024
#define NFP    256
#define NFR    256
#define NFPP   1280          // NFP + NFG
#define NGR    4
#define POS_TH2 (17.4f * 17.4f)             // (0.2*87)^2
#define SIM_SCALE 0.04419417382415922f      // 1/sqrt(NFR+NFP) = 1/sqrt(512)
#define LN_EPS 1e-5f

#define LDSTRIDE 40          // bf16 elements per LDS tile row (80 B pitch)

#if __has_builtin(__builtin_amdgcn_global_load_async_to_lds_b128) && \
    __has_builtin(__builtin_amdgcn_s_wait_asynccnt)
#define USE_ASYNC 1
typedef __attribute__((address_space(1))) v4i as1_v4i;   // global int4
typedef __attribute__((address_space(3))) v4i as3_v4i;   // LDS int4
#else
#define USE_ASYNC 0
#endif

// ----- WMMA fragment helpers ------------------------------------------------
// 16-bit A-matrix 16x32 layout (ISA 7.12.2): lane-half h = lane>>4 holds
// K = h*8 + {0..7} (elements 0..7) and K = h*8 + 16 + {0..7} (elements 8..15),
// row M = lane&15.  Same loader serves B given transposed (N x K row-major).
__device__ __forceinline__ v16bf load_frag_ld(const bf16_t* p) {
  v8bf lo = *(const v8bf*)(p);
  v8bf hi = *(const v8bf*)(p + 16);
  return __builtin_shufflevector(lo, hi, 0,1,2,3,4,5,6,7,8,9,10,11,12,13,14,15);
}

__device__ __forceinline__ v8f wmma_bf16(v16bf a, v16bf b, v8f c) {
  return __builtin_amdgcn_wmma_f32_16x16x32_bf16(false, a, false, b,
                                                 (short)0, c, false, false);
}

// Copy one 16-B chunk (chunk c: row c>>1, k-half c&1) of a tile into LDS.
__device__ __forceinline__ void stage_chunk(const bf16_t* __restrict__ G, int ld,
                                            int row0, int k0,
                                            bf16_t* lds, int c) {
  const int r = c >> 1;
  const int h = c & 1;         // two 8-element chunks: k = 0..7 and 16..23
  const bf16_t* src = G + (size_t)(row0 + r) * ld + k0 + h * 16;
  bf16_t* dst = lds + r * LDSTRIDE + h * 16;
#if USE_ASYNC
  __builtin_amdgcn_global_load_async_to_lds_b128(
      (as1_v4i*)(void*)src, (as3_v4i*)(void*)dst, 0, 0);
#else
  *(v8bf*)dst = *(const v8bf*)src;
#endif
}

// Stage a ROWSx32 bf16 tile; branch-free: compile-time chunk selection, so
// every async copy is unconditional (no divergent EXEC masking).
template <int ROWS>
__device__ __forceinline__ void stage_rows(const bf16_t* __restrict__ G, int ld,
                                           int row0, int k0,
                                           bf16_t* lds, int tid) {
  stage_chunk(G, ld, row0, k0, lds, tid);
  if (ROWS > 64)                       // compile-time constant
    stage_chunk(G, ld, row0, k0, lds, tid + 128);
}

// Block tile 128x64 (128 threads = 4 waves, each wave a 32x64 strip).
// Double-buffered LDS staging; acc0 rows wave*32+0..15, acc1 rows +16..31.
__device__ __forceinline__ void mma_tile_staged(const bf16_t* __restrict__ A,
                                                int lda,
                                                const bf16_t* __restrict__ Bt,
                                                int ldb,
                                                int blockRow, int blockCol,
                                                int K, v8f acc0[4], v8f acc1[4]) {
  __shared__ bf16_t As[2][128 * LDSTRIDE];
  __shared__ bf16_t Bs[2][64 * LDSTRIDE];
  const int tid  = threadIdx.x;
  const int lane = tid & 31;
  const int wave = tid >> 5;
  const int half = lane >> 4;
  const int rl   = lane & 15;
  const int nsteps = K >> 5;

  stage_rows<128>(A,  lda, blockRow, 0, As[0], tid);
  stage_rows<64>(Bt, ldb, blockCol, 0, Bs[0], tid);

  int cur = 0;
#pragma unroll 1
  for (int i = 0; i < nsteps; ++i) {
    if (i + 1 < nsteps) {
      stage_rows<128>(A,  lda, blockRow, (i + 1) * 32, As[cur ^ 1], tid);
      stage_rows<64>(Bt, ldb, blockCol, (i + 1) * 32, Bs[cur ^ 1], tid);
#if USE_ASYNC
      __builtin_amdgcn_s_wait_asynccnt(3);   // stage-i copies complete
#endif
    } else {
#if USE_ASYNC
      __builtin_amdgcn_s_wait_asynccnt(0);
#endif
    }
    __syncthreads();
    const bf16_t* Ab = As[cur];
    const bf16_t* Bb = Bs[cur];
    v16bf a0 = load_frag_ld(&Ab[(wave * 32 +      rl) * LDSTRIDE + half * 8]);
    v16bf a1 = load_frag_ld(&Ab[(wave * 32 + 16 + rl) * LDSTRIDE + half * 8]);
#pragma unroll
    for (int t = 0; t < 4; ++t) {
      v16bf b = load_frag_ld(&Bb[(t * 16 + rl) * LDSTRIDE + half * 8]);
      acc0[t] = wmma_bf16(a0, b, acc0[t]);
      acc1[t] = wmma_bf16(a1, b, acc1[t]);
    }
    __syncthreads();
    cur ^= 1;
  }
}

#define DECL_ACC(name) \
  v8f name[4] = {{0,0,0,0,0,0,0,0},{0,0,0,0,0,0,0,0}, \
                 {0,0,0,0,0,0,0,0},{0,0,0,0,0,0,0,0}}

// ----- prep kernels ---------------------------------------------------------

__global__ void k_zero(float* __restrict__ p, size_t n) {
  size_t i = (size_t)blockIdx.x * 256 + threadIdx.x;
  if (i < n) p[i] = 0.0f;
}

// fpp[n, 0:256] = sincos embedding of foot point, fpp[n, 256:1280] = feats.
// Also writes foot[n] = ((x1+x2)/2, y2).
__global__ void k_prep_fpp(const float* __restrict__ feats,
                           const float* __restrict__ boxes,
                           bf16_t* __restrict__ fpp,
                           float* __restrict__ foot) {
  int idx = blockIdx.x * 256 + threadIdx.x;      // over NBOX*NFPP
  int n = idx / NFPP;
  int d = idx - n * NFPP;
  float val;
  if (d < NFP) {
    float fx = (boxes[n * 4 + 0] + boxes[n * 4 + 2]) * 0.5f;
    float fy = boxes[n * 4 + 3];
    if (d == 0) { foot[n * 2 + 0] = fx; foot[n * 2 + 1] = fy; }
    int coord = d >> 7;           // 0 -> x, 1 -> y
    int dd    = d & 127;          // position inside 128-wide half
    int i     = dd & 63;
    float freq = __powf(10000.0f, -(float)i * (1.0f / 64.0f));
    float ang  = (coord ? fy : fx) * freq;
    val = (dd < 64) ? __sinf(ang) : __cosf(ang);
  } else {
    val = feats[(size_t)n * NFG + (d - NFP)];
  }
  fpp[idx] = (bf16_t)val;
}

// featsT[f, m] = feats[m, f]  (bf16)  -- B^T operand for the agg GEMM.
__global__ void k_prep_featsT(const float* __restrict__ feats,
                              bf16_t* __restrict__ featsT) {
  int idx = blockIdx.x * 256 + threadIdx.x;      // over NFG*NBOX
  int f = idx / NBOX;
  int m = idx - f * NBOX;
  featsT[idx] = (bf16_t)feats[(size_t)m * NFG + f];
}

// dst[g, n, k] = (bf16) src[g, k, n]  -- weight transpose+convert, per g (z).
__global__ void k_transpose_w(const float* __restrict__ src,
                              bf16_t* __restrict__ dst, int K, int N) {
  int g = blockIdx.z;
  size_t total = (size_t)K * N;
  size_t idx = (size_t)blockIdx.x * 256 + threadIdx.x;
  if (idx >= total) return;
  int n = (int)(idx / K);
  int k = (int)(idx - (size_t)n * K);
  dst[g * total + idx] = (bf16_t)src[g * total + (size_t)k * N + n];
}

// ----- GEMM kernels (128 threads = 4 waves, block tile 128x64) -------------

// D[g, n, r] = fpp[n, :] . Wt[g, r, :] + bias[g, r]   (-> bf16)
__global__ void k_gemm_bias_bf16(const bf16_t* __restrict__ A,
                                 const bf16_t* __restrict__ Wt,
                                 const float* __restrict__ bias,
                                 bf16_t* __restrict__ D) {
  const int lane = threadIdx.x & 31;
  const int wave = threadIdx.x >> 5;
  const int half = lane >> 4, rl = lane & 15;
  const int g    = blockIdx.z;
  const int row0 = blockIdx.y * 128 + wave * 32;
  const int col0 = blockIdx.x * 64;
  const bf16_t* Btg = Wt + (size_t)g * NFR * NFPP;
  DECL_ACC(acc0); DECL_ACC(acc1);
  mma_tile_staged(A, NFPP, Btg, NFPP, blockIdx.y * 128, col0, NFPP, acc0, acc1);
  bf16_t* Dg = D + (size_t)g * NBOX * NFR;
#pragma unroll
  for (int t = 0; t < 4; ++t) {
    int c = col0 + t * 16 + rl;
    float b = bias[g * NFR + c];
#pragma unroll
    for (int v = 0; v < 8; ++v) {
      int r = row0 + v + half * 8;
      Dg[(size_t)r * NFR + c]        = (bf16_t)(acc0[t][v] + b);
      Dg[(size_t)(r + 16) * NFR + c] = (bf16_t)(acc1[t][v] + b);
    }
  }
}

// S[n, m] = scale * theta[g,n,:] . phi[g,m,:]  masked by foot distance.
__global__ void k_gemm_sim(const bf16_t* __restrict__ theta,
                           const bf16_t* __restrict__ phi,
                           const float* __restrict__ foot,
                           float* __restrict__ S, int g) {
  const int lane = threadIdx.x & 31;
  const int wave = threadIdx.x >> 5;
  const int half = lane >> 4, rl = lane & 15;
  const int row0 = blockIdx.y * 128 + wave * 32;
  const int col0 = blockIdx.x * 64;
  const bf16_t* Ag = theta + (size_t)g * NBOX * NFR;
  const bf16_t* Bg = phi   + (size_t)g * NBOX * NFR;
  DECL_ACC(acc0); DECL_ACC(acc1);
  mma_tile_staged(Ag, NFR, Bg, NFR, blockIdx.y * 128, col0, NFR, acc0, acc1);
  float fnx[16], fny[16];
#pragma unroll
  for (int v = 0; v < 8; ++v) {
    int n = row0 + v + half * 8;
    fnx[v]     = foot[n * 2 + 0];
    fny[v]     = foot[n * 2 + 1];
    fnx[v + 8] = foot[(n + 16) * 2 + 0];
    fny[v + 8] = foot[(n + 16) * 2 + 1];
  }
#pragma unroll
  for (int t = 0; t < 4; ++t) {
    int m = col0 + t * 16 + rl;
    float mfx = foot[m * 2 + 0], mfy = foot[m * 2 + 1];
#pragma unroll
    for (int v = 0; v < 8; ++v) {
      int n = row0 + v + half * 8;
      float dx0 = fnx[v] - mfx, dy0 = fny[v] - mfy;
      float s0 = acc0[t][v] * SIM_SCALE;
      if (dx0 * dx0 + dy0 * dy0 > POS_TH2) s0 = -1.0e30f;
      S[(size_t)n * NBOX + m] = s0;
      float dx1 = fnx[v + 8] - mfx, dy1 = fny[v + 8] - mfy;
      float s1 = acc1[t][v] * SIM_SCALE;
      if (dx1 * dx1 + dy1 * dy1 > POS_TH2) s1 = -1.0e30f;
      S[(size_t)(n + 16) * NBOX + m] = s1;
    }
  }
}

// agg[n, f] = sum_m A[n, m] * feats[m, f]   (A bf16, featsT as B^T) -> bf16
__global__ void k_gemm_agg(const bf16_t* __restrict__ Abf,
                           const bf16_t* __restrict__ featsT,
                           bf16_t* __restrict__ agg) {
  const int lane = threadIdx.x & 31;
  const int wave = threadIdx.x >> 5;
  const int half = lane >> 4, rl = lane & 15;
  const int row0 = blockIdx.y * 128 + wave * 32;
  const int col0 = blockIdx.x * 64;
  DECL_ACC(acc0); DECL_ACC(acc1);
  mma_tile_staged(Abf, NBOX, featsT, NBOX, blockIdx.y * 128, col0, NBOX, acc0, acc1);
#pragma unroll
  for (int t = 0; t < 4; ++t) {
    int c = col0 + t * 16 + rl;
#pragma unroll
    for (int v = 0; v < 8; ++v) {
      int r = row0 + v + half * 8;
      agg[(size_t)r * NFG + c]        = (bf16_t)acc0[t][v];
      agg[(size_t)(r + 16) * NFG + c] = (bf16_t)acc1[t][v];
    }
  }
}

// O[n, o] = agg[n, :] . gcnWt[o, :]   (f32 out, LN done separately)
__global__ void k_gemm_out(const bf16_t* __restrict__ agg,
                           const bf16_t* __restrict__ gcnWt,
                           float* __restrict__ O) {
  const int lane = threadIdx.x & 31;
  const int wave = threadIdx.x >> 5;
  const int half = lane >> 4, rl = lane & 15;
  const int row0 = blockIdx.y * 128 + wave * 32;
  const int col0 = blockIdx.x * 64;
  DECL_ACC(acc0); DECL_ACC(acc1);
  mma_tile_staged(agg, NFG, gcnWt, NFG, blockIdx.y * 128, col0, NFG, acc0, acc1);
#pragma unroll
  for (int t = 0; t < 4; ++t) {
    int c = col0 + t * 16 + rl;
#pragma unroll
    for (int v = 0; v < 8; ++v) {
      int r = row0 + v + half * 8;
      O[(size_t)r * NFG + c]        = acc0[t][v];
      O[(size_t)(r + 16) * NFG + c] = acc1[t][v];
    }
  }
}

// ----- row softmax in LDS; writes bf16 A and accumulates A/NG into fg ------
__global__ void k_softmax(const float* __restrict__ S,
                          bf16_t* __restrict__ Abf,
                          float* __restrict__ fg) {
  __shared__ float buf[NBOX];      // 16 KB row cache
  __shared__ float red[256];
  const int row = blockIdx.x;
  const int t   = threadIdx.x;
  const float* srow = S + (size_t)row * NBOX;
  float lmax = -3.0e38f;
  for (int i = t; i < NBOX; i += 256) {
    float v = srow[i];
    buf[i] = v;
    lmax = fmaxf(lmax, v);
  }
  red[t] = lmax; __syncthreads();
  for (int s = 128; s > 0; s >>= 1) {
    if (t < s) red[t] = fmaxf(red[t], red[t + s]);
    __syncthreads();
  }
  float mx = red[0]; __syncthreads();
  float lsum = 0.0f;
  for (int i = t; i < NBOX; i += 256) {
    float e = __expf(buf[i] - mx);
    buf[i] = e;
    lsum += e;
  }
  red[t] = lsum; __syncthreads();
  for (int s = 128; s > 0; s >>= 1) {
    if (t < s) red[t] += red[t + s];
    __syncthreads();
  }
  float inv = 1.0f / red[0];
  for (int i = t; i < NBOX; i += 256) {
    float a = buf[i] * inv;
    Abf[(size_t)row * NBOX + i] = (bf16_t)a;
    fg[(size_t)row * NBOX + i] += 0.25f * a;   // mean over NG=4 graphs
  }
}

// ----- LayerNorm + gamma/beta + ReLU, accumulate over g --------------------
__global__ void k_ln_relu_acc(const float* __restrict__ O,
                              const float* __restrict__ gamma,
                              const float* __restrict__ beta,
                              float* __restrict__ out, int g) {
  __shared__ float rs[256], rq[256];
  const int row = blockIdx.x;
  const int t   = threadIdx.x;
  const float* orow = O + (size_t)row * NFG;
  float s = 0.0f, q = 0.0f;
  for (int i = t; i < NFG; i += 256) {
    float v = orow[i];
    s += v; q += v * v;
  }
  rs[t] = s; rq[t] = q; __syncthreads();
  for (int st = 128; st > 0; st >>= 1) {
    if (t < st) { rs[t] += rs[t + st]; rq[t] += rq[t + st]; }
    __syncthreads();
  }
  float mean = rs[0] * (1.0f / NFG);
  float var  = rq[0] * (1.0f / NFG) - mean * mean;
  float rstd = rsqrtf(var + LN_EPS);
  const float* gm = gamma + (size_t)g * NFG;
  const float* bt = beta  + (size_t)g * NFG;
  for (int i = t; i < NFG; i += 256) {
    float v = (orow[i] - mean) * rstd * gm[i] + bt[i];
    out[(size_t)row * NFG + i] += fmaxf(v, 0.0f);
  }
}

__global__ void k_clamp1(float* __restrict__ p, size_t n) {
  size_t i = (size_t)blockIdx.x * 256 + threadIdx.x;
  if (i < n) p[i] = fminf(p[i], 1.0f);
}

// ---------------------------------------------------------------------------
extern "C" void kernel_launch(void* const* d_in, const int* in_sizes, int n_in,
                              void* d_out, int out_size, void* d_ws, size_t ws_size,
                              hipStream_t stream) {
  (void)in_sizes; (void)n_in; (void)out_size; (void)ws_size;
  const float* feats   = (const float*)d_in[0];   // [1,4096,1024]
  const float* boxes   = (const float*)d_in[1];   // [4096,4]
  const float* theta_w = (const float*)d_in[2];   // [4,1280,256]
  const float* theta_b = (const float*)d_in[3];   // [4,256]
  const float* phi_w   = (const float*)d_in[4];
  const float* phi_b   = (const float*)d_in[5];
  const float* gcn_w   = (const float*)d_in[6];   // [4,1024,1024]
  const float* ln_g    = (const float*)d_in[7];
  const float* ln_b    = (const float*)d_in[8];

  float* out = (float*)d_out;                       // [4096,1024]
  float* fg  = out + (size_t)NBOX * NFG;            // [4096,4096]

  // --- workspace carve-out (256-byte aligned so 16-B vector loads are ok) --
  char* ws = (char*)d_ws;
  size_t off = 0;
  auto carve = [&](size_t bytes) -> char* {
    char* p = ws + off;
    off = (off + bytes + 255) & ~(size_t)255;
    return p;
  };
  float*  foot    = (float*) carve((size_t)NBOX * 2 * 4);
  bf16_t* fpp     = (bf16_t*)carve((size_t)NBOX * NFPP * 2);        // 10 MB
  bf16_t* featsT  = (bf16_t*)carve((size_t)NFG * NBOX * 2);         //  8 MB
  bf16_t* thetaWt = (bf16_t*)carve((size_t)NGR * NFR * NFPP * 2);   // 2.6 MB
  bf16_t* phiWt   = (bf16_t*)carve((size_t)NGR * NFR * NFPP * 2);
  bf16_t* gcnWt   = (bf16_t*)carve((size_t)NGR * NFG * NFG * 2);    //  8 MB
  bf16_t* theta   = (bf16_t*)carve((size_t)NGR * NBOX * NFR * 2);   //  8 MB
  bf16_t* phi     = (bf16_t*)carve((size_t)NGR * NBOX * NFR * 2);
  bf16_t* Abf     = (bf16_t*)carve((size_t)NBOX * NBOX * 2);        // 32 MB
  float*  Sbuf    = (float*) carve((size_t)NBOX * NBOX * 4);        // 64 MB
  bf16_t* agg     = (bf16_t*)carve((size_t)NBOX * NFG * 2);         //  8 MB
  float*  Obuf    = Sbuf;  // alias: sim scores dead once softmax consumed them

  dim3 th256(256), th128(128);

  size_t totOut = (size_t)NBOX * NFG + (size_t)NBOX * NBOX;
  k_zero<<<dim3((unsigned)((totOut + 255) / 256)), th256, 0, stream>>>(out, totOut);

  k_prep_fpp<<<dim3(NBOX * NFPP / 256), th256, 0, stream>>>(feats, boxes, fpp, foot);
  k_prep_featsT<<<dim3(NFG * NBOX / 256), th256, 0, stream>>>(feats, featsT);
  k_transpose_w<<<dim3(NFPP * NFR / 256, 1, NGR), th256, 0, stream>>>(theta_w, thetaWt, NFPP, NFR);
  k_transpose_w<<<dim3(NFPP * NFR / 256, 1, NGR), th256, 0, stream>>>(phi_w,   phiWt,   NFPP, NFR);
  k_transpose_w<<<dim3(NFG * NFG / 256, 1, NGR), th256, 0, stream>>>(gcn_w,   gcnWt,   NFG,  NFG);

  // theta/phi projections for all graphs (grid: cols/64, rows/128, g)
  k_gemm_bias_bf16<<<dim3(NFR / 64, NBOX / 128, NGR), th128, 0, stream>>>(fpp, thetaWt, theta_b, theta);
  k_gemm_bias_bf16<<<dim3(NFR / 64, NBOX / 128, NGR), th128, 0, stream>>>(fpp, phiWt,   phi_b,   phi);

  for (int g = 0; g < NGR; ++g) {
    k_gemm_sim<<<dim3(NBOX / 64, NBOX / 128), th128, 0, stream>>>(theta, phi, foot, Sbuf, g);
    k_softmax<<<dim3(NBOX), th256, 0, stream>>>(Sbuf, Abf, fg);
    k_gemm_agg<<<dim3(NFG / 64, NBOX / 128), th128, 0, stream>>>(Abf, featsT, agg);
    k_gemm_out<<<dim3(NFG / 64, NBOX / 128), th128, 0, stream>>>(agg, gcnWt + (size_t)g * NFG * NFG, Obuf);
    k_ln_relu_acc<<<dim3(NBOX), th256, 0, stream>>>(Obuf, ln_g, ln_b, out, g);
  }

  k_clamp1<<<dim3(NBOX * NBOX / 256), th256, 0, stream>>>(fg, (size_t)NBOX * NBOX);
}
